// GNN_60679297958153
// MI455X (gfx1250) — matrix operand order repaired and checked by reading, not compiled, not run on hardware
//
#include <hip/hip_runtime.h>
#include <math.h>

// ---------------------------------------------------------------------------
// CDNA5 (gfx1250) implementation of the hierarchical graph-transformer net.
// Dense linear layers run through V_WMMA_F32_16X16X32_BF16: each block stages
// a 32x64 bf16 B-tile in LDS in *fragment order* (Bs[col][krow]) so each
// lane's B fragment is 32 contiguous bytes -> 2x ds_load_b128.  Each wave
// computes a 16x64 output strip (4 WMMAs per K-step, A fragment reused 4x).
// All K%32==0 and N%64==0 for every GEMM in this net -> no bounds checks in
// the hot loop.  Edge attention / segment ops use f32 atomics +
// order-preserving uint atomicMax for segment-max.
// ---------------------------------------------------------------------------

typedef __attribute__((ext_vector_type(16))) __bf16 v16bf;
typedef __attribute__((ext_vector_type(2)))  __bf16 v2bf;
typedef __attribute__((ext_vector_type(8)))  float  v8f;
typedef __attribute__((ext_vector_type(4)))  float  v4f;

#define HEADS 3
#define CH    64
#define HC    192     /* HEADS*CH */
#define HIDD  64
#define GNUM  500

__device__ __forceinline__ unsigned fenc(float f) {
  unsigned u = __float_as_uint(f);
  return (u & 0x80000000u) ? ~u : (u | 0x80000000u);
}
__device__ __forceinline__ float fdec(unsigned u) {
  return __uint_as_float((u & 0x80000000u) ? (u & 0x7FFFFFFFu) : ~u);
}

// ------------------------------ fills --------------------------------------
__global__ void k_fill_f32(float* p, float v, long long n) {
  long long i = (long long)blockIdx.x * blockDim.x + threadIdx.x;
  if (i < n) p[i] = v;
}
__global__ void k_fill_u32(unsigned* p, unsigned v, long long n) {
  long long i = (long long)blockIdx.x * blockDim.x + threadIdx.x;
  if (i < n) p[i] = v;
}
__global__ void k_fill_u8(unsigned char* p, unsigned char v, long long n) {
  long long i = (long long)blockIdx.x * blockDim.x + threadIdx.x;
  if (i < n) p[i] = v;
}
__global__ void k_axpy(float* acc, const float* x, long long n) {
  long long i = (long long)blockIdx.x * blockDim.x + threadIdx.x;
  if (i < n) acc[i] += x[i];
}

// ------------------------------ WMMA GEMM ----------------------------------
// out[M,N] = relu?( A[M,K] @ W[K,N] + bias[N] ).  Requires K%32==0, N%64==0.
// Block: 256 threads = 8 waves.  Block tile: 128 rows x 64 cols.
// Wave w: rows [bx*128 + w*16, +16), all 64 cols (4 accumulators).
// LDS B tile is fragment-ordered: Bs[col][krow], so a lane's fragment
// (16 consecutive krows for one col) is one contiguous 32B LDS read.
__global__ __launch_bounds__(256) void k_gemm_bf16(
    const float* __restrict__ A, const float* __restrict__ W,
    const float* __restrict__ bias, float* __restrict__ out,
    int M, int K, int N, int relu)
{
  __shared__ __bf16 Bs[64 * 32];                     // [col][krow]
  const int tid  = threadIdx.x;
  const int lane = tid & 31;
  const int wave = tid >> 5;
  const int l15  = lane & 15;
  const int hi   = lane >> 4;
  const int row  = blockIdx.x * 128 + wave * 16 + l15;
  const int rowc = (row < M) ? row : (M - 1);        // clamp: loads stay valid
  const int nc0  = blockIdx.y * 64;

  const int krp = tid >> 4;                          // 0..15 -> krows 2k,2k+1
  const int cg4 = (tid & 15) << 2;                   // col group of 4

  v8f acc[4];
  acc[0] = (v8f){}; acc[1] = (v8f){}; acc[2] = (v8f){}; acc[3] = (v8f){};

  for (int k0 = 0; k0 < K; k0 += 32) {
    // ---- cooperative B tile staging (fragment order) ----
    // thread: rows k0+2*krp, k0+2*krp+1; cols nc0+cg4 .. +4
    {
      const float* w0 = W + (size_t)(k0 + 2 * krp) * N + nc0 + cg4;
      v4f r0 = *(const v4f*)w0;                      // global b128
      v4f r1 = *(const v4f*)(w0 + N);                // global b128
#pragma unroll
      for (int j = 0; j < 4; ++j) {
        v2bf pk;
        pk[0] = (__bf16)r0[j];                       // krow 2*krp
        pk[1] = (__bf16)r1[j];                       // krow 2*krp+1
        *(v2bf*)&Bs[(cg4 + j) * 32 + 2 * krp] = pk;  // 4B aligned LDS store
      }
    }
    __syncthreads();
    // ---- A fragment: af[i] holds K = hi*16 + i ----
    v16bf af;
    {
      const v4f* ap = (const v4f*)(A + (size_t)rowc * K + k0 + (hi << 4));
      v4f x0 = ap[0], x1 = ap[1], x2 = ap[2], x3 = ap[3];
#pragma unroll
      for (int j = 0; j < 4; ++j) {
        af[j]      = (__bf16)x0[j];
        af[4 + j]  = (__bf16)x1[j];
        af[8 + j]  = (__bf16)x2[j];
        af[12 + j] = (__bf16)x3[j];
      }
    }
    // ---- 4 WMMAs, B fragments = contiguous 32B LDS reads ----
#pragma unroll
    for (int t = 0; t < 4; ++t) {
      // lane's col = t*16 + l15; krows hi*16 .. hi*16+15 contiguous
      v16bf bf = *(const v16bf*)&Bs[(t * 16 + l15) * 32 + (hi << 4)];
      acc[t] = __builtin_amdgcn_wmma_f32_16x16x32_bf16(false, af, false, bf,
                                                       (short)0, acc[t],
                                                       false, false);
    }
    __syncthreads();
  }
  // ---- store: C/D layout VGPR r -> M=r (lanes16-31: +8), N=lane&15 ----
#pragma unroll
  for (int t = 0; t < 4; ++t) {
    int col = nc0 + t * 16 + l15;
    float bb = (bias != nullptr) ? bias[col] : 0.0f;
#pragma unroll
    for (int rr = 0; rr < 8; ++rr) {
      int orow = blockIdx.x * 128 + wave * 16 + rr + (hi << 3);
      if (orow < M) {
        float v = acc[t][rr] + bb;
        if (relu) v = fmaxf(v, 0.0f);
        out[(size_t)orow * N + col] = v;
      }
    }
  }
}

// --------------------------- edge attention --------------------------------
// a[e,h] = em ? dot(q[dst,h,:], k[src,h,:] + e[e,h,:]) / 8 : -inf
// edge projection e[e,h,c] recomputed on the fly from (ea1|ea2) @ We.
// K1/K2 are compile-time: edge-attr row lives in registers, dots unroll.
template <int K1, int K2>
__global__ void k_edge_attn(const float* __restrict__ q,
                            const float* __restrict__ kk,
                            const float* __restrict__ ea1,
                            const float* __restrict__ ea2,
                            const float* __restrict__ We,
                            const int* __restrict__ src,
                            const int* __restrict__ dst,
                            const unsigned char* __restrict__ em,
                            float* __restrict__ a, int E)
{
  long long idx = (long long)blockIdx.x * blockDim.x + threadIdx.x;
  if (idx >= (long long)E * HEADS) return;
  int e = (int)(idx / HEADS), h = (int)(idx % HEADS);
  if (!em[e]) { a[idx] = -__builtin_inff(); return; }
  int s = src[e], d = dst[e];
  float er[K1 + K2];
#pragma unroll
  for (int t = 0; t < K1; ++t) er[t] = ea1[(size_t)e * K1 + t];
#pragma unroll
  for (int t = 0; t < K2; ++t) er[K1 + t] = ea2[(size_t)e * K2 + t];
  const float* qr = &q[(size_t)d * HC + h * CH];
  const float* kr = &kk[(size_t)s * HC + h * CH];
  float acc = 0.0f;
  for (int c = 0; c < CH; ++c) {
    float ec = 0.0f;
#pragma unroll
    for (int t = 0; t < K1 + K2; ++t)
      ec += er[t] * We[(size_t)t * HC + h * CH + c];
    acc += qr[c] * (kr[c] + ec);
  }
  a[idx] = acc * 0.125f;                    // 1/sqrt(64)
}

__global__ void k_segmax(const float* __restrict__ a, const int* __restrict__ dst,
                         unsigned* __restrict__ mkey, int E) {
  long long idx = (long long)blockIdx.x * blockDim.x + threadIdx.x;
  if (idx >= (long long)E * HEADS) return;
  float v = a[idx];
  if (!__builtin_isfinite(v)) return;
  int e = (int)(idx / HEADS), h = (int)(idx % HEADS);
  atomicMax(&mkey[(size_t)dst[e] * HEADS + h], fenc(v));
}

__global__ void k_expsum(float* __restrict__ a, const int* __restrict__ dst,
                         const unsigned* __restrict__ mkey,
                         float* __restrict__ s, int E) {
  long long idx = (long long)blockIdx.x * blockDim.x + threadIdx.x;
  if (idx >= (long long)E * HEADS) return;
  float v = a[idx];
  if (!__builtin_isfinite(v)) { a[idx] = 0.0f; return; }
  int e = (int)(idx / HEADS), h = (int)(idx % HEADS);
  float m = fdec(mkey[(size_t)dst[e] * HEADS + h]);
  if (!__builtin_isfinite(m)) m = 0.0f;
  float ex = expf(v - m);
  a[idx] = ex;
  atomicAdd(&s[(size_t)dst[e] * HEADS + h], ex);
}

template <int K1, int K2>
__global__ void k_edgeout(const float* __restrict__ exb,
                          const float* __restrict__ v,
                          const float* __restrict__ ea1,
                          const float* __restrict__ ea2,
                          const float* __restrict__ We,
                          const int* __restrict__ src,
                          const int* __restrict__ dst,
                          const float* __restrict__ s,
                          float* __restrict__ outb, int E)
{
  long long idx = (long long)blockIdx.x * blockDim.x + threadIdx.x;
  if (idx >= (long long)E * HEADS) return;
  float ex = exb[idx];
  if (ex == 0.0f) return;
  int e = (int)(idx / HEADS), h = (int)(idx % HEADS);
  int sn = src[e], d = dst[e];
  float sd = s[(size_t)d * HEADS + h];
  float alpha = ex / (sd > 0.0f ? sd : 1.0f);
  float er[K1 + K2];
#pragma unroll
  for (int t = 0; t < K1; ++t) er[t] = ea1[(size_t)e * K1 + t];
#pragma unroll
  for (int t = 0; t < K2; ++t) er[K1 + t] = ea2[(size_t)e * K2 + t];
  const float* vr = &v[(size_t)sn * HC + h * CH];
  float* orow = &outb[(size_t)d * HC + h * CH];
  for (int c = 0; c < CH; ++c) {
    float ec = 0.0f;
#pragma unroll
    for (int t = 0; t < K1 + K2; ++t)
      ec += er[t] * We[(size_t)t * HC + h * CH + c];
    atomicAdd(&orow[c], (vr[c] + ec) * alpha);
  }
}

// beta gate: b = sigmoid([out, xr, out-xr] @ bw); x = b*xr + (1-b)*out
__global__ void k_beta(float* __restrict__ outb, const float* __restrict__ xr,
                       const float* __restrict__ bw, int n) {
  int i = blockIdx.x * blockDim.x + threadIdx.x;
  if (i >= n) return;
  float acc = 0.0f;
  for (int j = 0; j < HC; ++j) {
    float o = outb[(size_t)i * HC + j], r = xr[(size_t)i * HC + j];
    acc += o * bw[j] + r * bw[HC + j] + (o - r) * bw[2 * HC + j];
  }
  float b = 1.0f / (1.0f + expf(-acc));
  for (int j = 0; j < HC; ++j) {
    float o = outb[(size_t)i * HC + j], r = xr[(size_t)i * HC + j];
    outb[(size_t)i * HC + j] = b * r + (1.0f - b) * o;
  }
}

// ------------------------------ batch norm ---------------------------------
__global__ void k_bn_stats(const float* __restrict__ x,
                           const unsigned char* __restrict__ nm,
                           float* __restrict__ st, int n) {
  int c = threadIdx.x;                      // blockDim = 64
  int start = blockIdx.x * 512;
  int end = min(n, start + 512);
  float s = 0.0f, sq = 0.0f; int cl = 0;
  for (int i = start; i < end; ++i) {
    float w = nm[i] ? 1.0f : 0.0f;
    float v = x[(size_t)i * HIDD + c];
    s += w * v; sq += w * v * v;
    cl += nm[i] ? 1 : 0;
  }
  atomicAdd(&st[c], s);
  atomicAdd(&st[HIDD + c], sq);
  if (c == 0) atomicAdd(&st[2 * HIDD], (float)cl);
}

__global__ void k_bn_apply(const float* __restrict__ x, float* __restrict__ y,
                           const float* __restrict__ st,
                           const float* __restrict__ g,
                           const float* __restrict__ b, int n) {
  long long idx = (long long)blockIdx.x * blockDim.x + threadIdx.x;
  if (idx >= (long long)n * HIDD) return;
  int c = (int)(idx % HIDD);
  float cnt = fmaxf(st[2 * HIDD], 1.0f);
  float mu = st[c] / cnt;
  float var = fmaxf(st[HIDD + c] / cnt - mu * mu, 0.0f);
  y[idx] = (x[idx] - mu) * rsqrtf(var + 1e-5f) * g[c] + b[c];
}

// ------------------------------ misc ---------------------------------------
__global__ void k_concat2(const float* a, const float* b, float* o, int n) {
  long long idx = (long long)blockIdx.x * blockDim.x + threadIdx.x;
  if (idx >= (long long)n * HIDD) return;
  int i = (int)(idx / HIDD), c = (int)(idx % HIDD);
  o[(size_t)i * 128 + c] = a[idx];
  o[(size_t)i * 128 + HIDD + c] = b[idx];
}
__global__ void k_maskmul(float* x, const unsigned char* nm, int n) {
  long long idx = (long long)blockIdx.x * blockDim.x + threadIdx.x;
  if (idx >= (long long)n * HIDD) return;
  if (!nm[idx / HIDD]) x[idx] = 0.0f;
}

// ------------------------------ SAG pooling --------------------------------
__global__ void k_sag_agg(const float* __restrict__ x, const int* __restrict__ src,
                          const int* __restrict__ dst,
                          const unsigned char* __restrict__ em,
                          float* __restrict__ agg, int E) {
  int e = blockIdx.x * blockDim.x + threadIdx.x;
  if (e >= E || !em[e]) return;
  int s = src[e], d = dst[e];
  for (int c = 0; c < HIDD; ++c)
    atomicAdd(&agg[(size_t)d * HIDD + c], x[(size_t)s * HIDD + c]);
}
__global__ void k_sag_score(const float* __restrict__ x, const float* __restrict__ agg,
                            const float* __restrict__ relw, const float* __restrict__ relb,
                            const float* __restrict__ rootw,
                            float* __restrict__ score, int n) {
  int i = blockIdx.x * blockDim.x + threadIdx.x;
  if (i >= n) return;
  float acc = relb[0];
  for (int c = 0; c < HIDD; ++c)
    acc += agg[(size_t)i * HIDD + c] * relw[c] + x[(size_t)i * HIDD + c] * rootw[c];
  score[i] = tanhf(acc);
}
__global__ void k_counts(const int* __restrict__ batch,
                         const unsigned char* __restrict__ nm,
                         int* total, int* cnt, int n) {
  int i = blockIdx.x * blockDim.x + threadIdx.x;
  if (i >= n) return;
  atomicAdd(&total[batch[i]], 1);
  if (nm[i]) atomicAdd(&cnt[batch[i]], 1);
}
__global__ void k_scan_starts(const int* total, int* starts, int Gn) {
  __shared__ int sh[512];
  int t = threadIdx.x;
  sh[t] = (t < Gn) ? total[t] : 0;
  __syncthreads();
  for (int off = 1; off < 512; off <<= 1) {
    int v = (t >= off) ? sh[t - off] : 0;
    __syncthreads();
    sh[t] += v;
    __syncthreads();
  }
  if (t < Gn) starts[t] = sh[t] - total[t];
}
// one block per graph: exact stable rank (lexsort by (batch, -score))
__global__ void k_sag_rank(const float* __restrict__ score,
                           const unsigned char* __restrict__ nm,
                           const int* __restrict__ starts,
                           const int* __restrict__ total,
                           const int* __restrict__ cnt,
                           const float* __restrict__ x,
                           float* __restrict__ xn,
                           unsigned char* __restrict__ keep, int Gn) {
  int g = blockIdx.x;
  int s0 = starts[g], tn = total[g];
  int kk = (int)ceilf(0.8f * (float)cnt[g]);
  for (int ii = threadIdx.x; ii < tn; ii += blockDim.x) {
    int i = s0 + ii;
    float si = nm[i] ? score[i] : -__builtin_inff();
    int rank = 0;
    for (int jj = 0; jj < tn; ++jj) {
      int j = s0 + jj;
      float sj = nm[j] ? score[j] : -__builtin_inff();
      if (sj > si || (sj == si && jj < ii)) rank++;
    }
    bool kp = nm[i] && (rank < kk);
    keep[i] = kp ? 1 : 0;
    float f = kp ? score[i] : 0.0f;
    for (int c = 0; c < HIDD; ++c)
      xn[(size_t)i * HIDD + c] = x[(size_t)i * HIDD + c] * f;
  }
}
__global__ void k_edgemask(unsigned char* em, const unsigned char* keep,
                           const int* src, const int* dst, int E) {
  int e = blockIdx.x * blockDim.x + threadIdx.x;
  if (e >= E) return;
  em[e] = (em[e] && keep[src[e]] && keep[dst[e]]) ? 1 : 0;
}

// ------------------------------ global pools -------------------------------
__global__ void k_pool_acc(const float* __restrict__ x, const int* __restrict__ batch,
                           const unsigned char* __restrict__ nm,
                           unsigned* __restrict__ mxkey, float* __restrict__ sm,
                           float* __restrict__ cntf, int n) {
  int i = blockIdx.x * blockDim.x + threadIdx.x;
  if (i >= n || !nm[i]) return;
  int g = batch[i];
  atomicAdd(&cntf[g], 1.0f);
  for (int c = 0; c < HIDD; ++c) {
    float v = x[(size_t)i * HIDD + c];
    atomicAdd(&sm[(size_t)g * HIDD + c], v);
    atomicMax(&mxkey[(size_t)g * HIDD + c], fenc(v));
  }
}
__global__ void k_pool_emit(const unsigned* mxkey, const float* sm,
                            const float* cntf, float* hg, int Gn) {
  int idx = blockIdx.x * blockDim.x + threadIdx.x;
  if (idx >= Gn * HIDD) return;
  int g = idx / HIDD, c = idx % HIDD;
  float mx = fdec(mxkey[idx]);
  if (!__builtin_isfinite(mx)) mx = 0.0f;
  float sv = sm[idx];
  float cn = fmaxf(cntf[g], 1.0f);
  hg[(size_t)g * HC + c]            += mx;
  hg[(size_t)g * HC + HIDD + c]     += sv / cn;
  hg[(size_t)g * HC + 2 * HIDD + c] += sv;
}

// ---------------------------------------------------------------------------
// Host-side orchestration
// ---------------------------------------------------------------------------
namespace {
struct ArgReader {
  void* const* p; int i;
  const float* f() { return (const float*)p[i++]; }
  const int*   d() { return (const int*)p[i++]; }
};
struct TCW { const float *qw,*qb,*kw,*kb,*vw,*vb,*ew,*skw,*skb,*bw; };
struct DW  { const float *w, *b; };
struct BNW { const float *g, *b; };
struct PW  { const float *relw, *relb, *rootw; };
static TCW rd_tc(ArgReader& r) {
  TCW t;
  t.qw = r.f(); t.qb = r.f(); t.kw = r.f(); t.kb = r.f();
  t.vw = r.f(); t.vb = r.f(); t.ew = r.f();
  t.skw = r.f(); t.skb = r.f(); t.bw = r.f();
  return t;
}
static DW  rd_dn(ArgReader& r) { DW d; d.w = r.f(); d.b = r.f(); return d; }
static BNW rd_bn(ArgReader& r) { BNW b; b.g = r.f(); b.b = r.f(); return b; }
static PW  rd_pl(ArgReader& r) { PW p; p.relw = r.f(); p.relb = r.f(); p.rootw = r.f(); return p; }
static inline unsigned nb(long long c) { return (unsigned)((c + 255) / 256); }
} // namespace

extern "C" void kernel_launch(void* const* d_in, const int* in_sizes, int n_in,
                              void* d_out, int out_size, void* d_ws, size_t ws_size,
                              hipStream_t stream)
{
  // ---- parse params in setup_inputs() insertion order ----
  ArgReader r{d_in, 0};
  TCW tc2d1 = rd_tc(r); DW emb2d1 = rd_dn(r); BNW bn2d1 = rd_bn(r);
  TCW tc3d1 = rd_tc(r); DW emb3d1 = rd_dn(r); BNW bn3d1 = rd_bn(r);
  TCW tc2d2[3]; for (int i = 0; i < 3; ++i) tc2d2[i] = rd_tc(r);
  DW  emb2d2[3]; for (int i = 0; i < 3; ++i) emb2d2[i] = rd_dn(r);
  BNW bn2d2[3];  for (int i = 0; i < 3; ++i) bn2d2[i] = rd_bn(r);
  TCW tc3d2[3]; for (int i = 0; i < 3; ++i) tc3d2[i] = rd_tc(r);
  DW  emb3d2[3]; for (int i = 0; i < 3; ++i) emb3d2[i] = rd_dn(r);
  BNW bn3d2[3];  for (int i = 0; i < 3; ++i) bn3d2[i] = rd_bn(r);
  TCW tc3 = rd_tc(r); DW emb3 = rd_dn(r);
  TCW tc4[3]; for (int i = 0; i < 3; ++i) tc4[i] = rd_tc(r);
  DW  emb4[3]; for (int i = 0; i < 3; ++i) emb4[i] = rd_dn(r);
  BNW bn4[3];  for (int i = 0; i < 3; ++i) bn4[i] = rd_bn(r);
  PW  pool[3]; for (int i = 0; i < 3; ++i) pool[i] = rd_pl(r);
  DW fc1 = rd_dn(r), fc2 = rd_dn(r);
  int i_n2d = r.i;  const float* n2d = r.f();
  const float* n3d = r.f();
  int i_ea2d = r.i; const float* ea2d = r.f();
  const float* ea3d = r.f();
  const int* eidx  = r.d();
  const int* batch = r.d();

  const int n = in_sizes[i_n2d] / 64;
  const int E = in_sizes[i_ea2d] / 16;
  const int Gn = GNUM;
  const int* src = eidx;
  const int* dst = eidx + E;

  // ---- workspace bump allocator ----
  char* base = (char*)d_ws; size_t off = 0;
  auto alloc = [&](size_t bytes) -> void* {
    void* p = base + off; off += (bytes + 255) & ~(size_t)255; return p;
  };
  float* q    = (float*)alloc((size_t)n * HC * 4);
  float* kbf  = (float*)alloc((size_t)n * HC * 4);
  float* vbf  = (float*)alloc((size_t)n * HC * 4);
  float* xr   = (float*)alloc((size_t)n * HC * 4);
  float* outb = (float*)alloc((size_t)n * HC * 4);
  float* embt = (float*)alloc((size_t)n * HIDD * 4);
  float* abuf = (float*)alloc((size_t)E * HEADS * 4);
  unsigned* mkey = (unsigned*)alloc((size_t)n * HEADS * 4);
  float* sseg = (float*)alloc((size_t)n * HEADS * 4);
  float* bnst = (float*)alloc(256 * 4);
  float* nodeA = (float*)alloc((size_t)n * HIDD * 4);
  float* nodeB = (float*)alloc((size_t)n * HIDD * 4);
  float* accA  = (float*)alloc((size_t)n * HIDD * 4);
  float* accB  = (float*)alloc((size_t)n * HIDD * 4);
  float* xcat  = (float*)alloc((size_t)n * 128 * 4);
  float* aggb  = (float*)alloc((size_t)n * HIDD * 4);
  float* score = (float*)alloc((size_t)n * 4);
  unsigned char* nmA = (unsigned char*)alloc(n);
  unsigned char* nmB = (unsigned char*)alloc(n);
  unsigned char* em  = (unsigned char*)alloc(E);
  int* total  = (int*)alloc(Gn * 4);
  int* cntg   = (int*)alloc(Gn * 4);
  int* starts = (int*)alloc(Gn * 4);
  unsigned* mxkey = (unsigned*)alloc((size_t)Gn * HIDD * 4);
  float* smp  = (float*)alloc((size_t)Gn * HIDD * 4);
  float* cntf = (float*)alloc(Gn * 4);
  float* hg   = (float*)alloc((size_t)Gn * HC * 4);
  float* fctmp = (float*)alloc((size_t)Gn * 256 * 4);
  (void)ws_size; (void)n_in; (void)out_size;

  auto fillf = [&](float* p, float v, long long c) {
    k_fill_f32<<<dim3(nb(c)), 256, 0, stream>>>(p, v, c);
  };
  auto gemm = [&](const float* A, const float* W, const float* b, float* o,
                  int M, int K, int N, int relu) {
    dim3 g((M + 127) / 128, (N + 63) / 64);
    k_gemm_bf16<<<g, 256, 0, stream>>>(A, W, b, o, M, K, N, relu);
  };
  auto tconv = [&](const TCW& p, const float* x, int din,
                   const float* ea1, int K1, const float* ea2, int K2) {
    gemm(x, p.qw, p.qb, q,   n, din, HC, 0);
    gemm(x, p.kw, p.kb, kbf, n, din, HC, 0);
    gemm(x, p.vw, p.vb, vbf, n, din, HC, 0);
    gemm(x, p.skw, p.skb, xr, n, din, HC, 0);
    long long eh = (long long)E * HEADS;
    dim3 ge(nb(eh));
    if (K1 == 16 && K2 == 0)
      k_edge_attn<16, 0><<<ge, 256, 0, stream>>>(q, kbf, ea1, ea2, p.ew, src, dst, em, abuf, E);
    else if (K1 == 8 && K2 == 0)
      k_edge_attn<8, 0><<<ge, 256, 0, stream>>>(q, kbf, ea1, ea2, p.ew, src, dst, em, abuf, E);
    else
      k_edge_attn<16, 8><<<ge, 256, 0, stream>>>(q, kbf, ea1, ea2, p.ew, src, dst, em, abuf, E);
    k_fill_u32<<<dim3(nb((long long)n * HEADS)), 256, 0, stream>>>(mkey, 0u, (long long)n * HEADS);
    k_segmax<<<ge, 256, 0, stream>>>(abuf, dst, mkey, E);
    fillf(sseg, 0.0f, (long long)n * HEADS);
    k_expsum<<<ge, 256, 0, stream>>>(abuf, dst, mkey, sseg, E);
    fillf(outb, 0.0f, (long long)n * HC);
    if (K1 == 16 && K2 == 0)
      k_edgeout<16, 0><<<ge, 256, 0, stream>>>(abuf, vbf, ea1, ea2, p.ew, src, dst, sseg, outb, E);
    else if (K1 == 8 && K2 == 0)
      k_edgeout<8, 0><<<ge, 256, 0, stream>>>(abuf, vbf, ea1, ea2, p.ew, src, dst, sseg, outb, E);
    else
      k_edgeout<16, 8><<<ge, 256, 0, stream>>>(abuf, vbf, ea1, ea2, p.ew, src, dst, sseg, outb, E);
    k_beta<<<dim3(nb(n)), 256, 0, stream>>>(outb, xr, p.bw, n);   // result in outb
  };
  auto blockf = [&](const TCW& p, const DW& e, const BNW& b, const float* x,
                    int din, const float* ea1, int K1, const float* ea2, int K2,
                    const unsigned char* nm, float* xout) {
    tconv(p, x, din, ea1, K1, ea2, K2);
    gemm(outb, e.w, e.b, embt, n, HC, HIDD, 1);                   // relu
    fillf(bnst, 0.0f, 2 * HIDD + 1);
    k_bn_stats<<<dim3((n + 511) / 512), 64, 0, stream>>>(embt, nm, bnst, n);
    k_bn_apply<<<dim3(nb((long long)n * HIDD)), 256, 0, stream>>>(embt, xout, bnst, b.g, b.b, n);
  };

  // ---- masks ----
  k_fill_u8<<<dim3(nb(n)), 256, 0, stream>>>(nmA, 1, n);
  k_fill_u8<<<dim3(nb(E)), 256, 0, stream>>>(em, 1, E);

  // ---- 2D branch ----
  blockf(tc2d1, emb2d1, bn2d1, n2d, 64, ea2d, 16, nullptr, 0, nmA, nodeA);
  fillf(accA, 0.0f, (long long)n * HIDD);
  for (int i = 0; i < 3; ++i) {
    blockf(tc2d2[i], emb2d2[i], bn2d2[i], nodeA, HIDD, ea2d, 16, nullptr, 0, nmA, nodeA);
    k_axpy<<<dim3(nb((long long)n * HIDD)), 256, 0, stream>>>(accA, nodeA, (long long)n * HIDD);
  }
  // ---- 3D branch ----
  blockf(tc3d1, emb3d1, bn3d1, n3d, 64, ea3d, 8, nullptr, 0, nmA, nodeB);
  fillf(accB, 0.0f, (long long)n * HIDD);
  for (int i = 0; i < 3; ++i) {
    blockf(tc3d2[i], emb3d2[i], bn3d2[i], nodeB, HIDD, ea3d, 8, nullptr, 0, nmA, nodeB);
    k_axpy<<<dim3(nb((long long)n * HIDD)), 256, 0, stream>>>(accB, nodeB, (long long)n * HIDD);
  }
  // ---- fuse branches ----
  k_concat2<<<dim3(nb((long long)n * HIDD)), 256, 0, stream>>>(accA, accB, xcat, n);
  tconv(tc3, xcat, 128, ea2d, 16, ea3d, 8);
  gemm(outb, emb3.w, emb3.b, nodeA, n, HC, HIDD, 0);              // no relu

  // ---- tc4 + SAGPool + pools ----
  fillf(hg, 0.0f, (long long)Gn * HC);
  float* xcur = nodeA; float* xalt = nodeB;
  unsigned char* nmcur = nmA; unsigned char* nmalt = nmB;
  for (int i = 0; i < 3; ++i) {
    blockf(tc4[i], emb4[i], bn4[i], xcur, HIDD, ea2d, 16, ea3d, 8, nmcur, xcur);
    k_maskmul<<<dim3(nb((long long)n * HIDD)), 256, 0, stream>>>(xcur, nmcur, n);
    // SAGPooling
    fillf(aggb, 0.0f, (long long)n * HIDD);
    k_sag_agg<<<dim3(nb(E)), 256, 0, stream>>>(xcur, src, dst, em, aggb, E);
    k_sag_score<<<dim3(nb(n)), 256, 0, stream>>>(xcur, aggb, pool[i].relw,
                                                 pool[i].relb, pool[i].rootw, score, n);
    k_fill_u32<<<dim3(nb(Gn)), 256, 0, stream>>>((unsigned*)total, 0u, Gn);
    k_fill_u32<<<dim3(nb(Gn)), 256, 0, stream>>>((unsigned*)cntg, 0u, Gn);
    k_counts<<<dim3(nb(n)), 256, 0, stream>>>(batch, nmcur, total, cntg, n);
    k_scan_starts<<<1, 512, 0, stream>>>(total, starts, Gn);
    k_sag_rank<<<Gn, 128, 0, stream>>>(score, nmcur, starts, total, cntg, xcur, xalt, nmalt, Gn);
    k_edgemask<<<dim3(nb(E)), 256, 0, stream>>>(em, nmalt, src, dst, E);
    { float* t = xcur; xcur = xalt; xalt = t; }
    { unsigned char* t = nmcur; nmcur = nmalt; nmalt = t; }
    // global max/mean/add pools -> hg
    k_fill_u32<<<dim3(nb((long long)Gn * HIDD)), 256, 0, stream>>>(mxkey, 0u, (long long)Gn * HIDD);
    fillf(smp, 0.0f, (long long)Gn * HIDD);
    fillf(cntf, 0.0f, Gn);
    k_pool_acc<<<dim3(nb(n)), 256, 0, stream>>>(xcur, batch, nmcur, mxkey, smp, cntf, n);
    k_pool_emit<<<dim3(nb((long long)Gn * HIDD)), 256, 0, stream>>>(mxkey, smp, cntf, hg, Gn);
  }

  // ---- FC head (WMMA) ----
  gemm(hg, fc1.w, fc1.b, fctmp, Gn, HC, 256, 1);
  gemm(fctmp, fc2.w, fc2.b, (float*)d_out, Gn, 256, 256, 1);
}
